// HeadAttention_63247688401413
// MI455X (gfx1250) — matrix operand order repaired and checked by reading, not compiled, not run on hardware
//
#include <hip/hip_runtime.h>
#include <hip/hip_bf16.h>

typedef __attribute__((ext_vector_type(16))) _Float16 v16h;
typedef __attribute__((ext_vector_type(8)))  float    v8f;

#define BATCH 4
#define SEQ   4096
#define CDIM  1024
#define HDIM  64
#define MROWS (BATCH * SEQ)   // 16384

// ---------------------------------------------------------------------------
// Kernel 0: convert W [C][H] fp32 -> Wt [H][C] f16 (Wq scaled by 1/sqrt(H))
// ---------------------------------------------------------------------------
__global__ void prep_w(const float* __restrict__ Wq, const float* __restrict__ Wk,
                       const float* __restrict__ Wv, _Float16* __restrict__ WT) {
  int idx = blockIdx.x * blockDim.x + threadIdx.x;
  if (idx >= 3 * CDIM * HDIM) return;
  int mat = idx / (CDIM * HDIM);
  int rem = idx % (CDIM * HDIM);
  int c = rem / HDIM, h = rem % HDIM;
  const float* W = (mat == 0) ? Wq : (mat == 1 ? Wk : Wv);
  float val = W[c * HDIM + h];
  if (mat == 0) val *= 0.125f;  // fold softmax scale 1/sqrt(64) into Q
  WT[(size_t)mat * HDIM * CDIM + (size_t)h * CDIM + c] = (_Float16)val;
}

// ---------------------------------------------------------------------------
// Kernel 1: QKV projection.  [16384 x 1024] x [1024 x 64] for q,k,v.
// One wave computes a 16x64 strip: 1 A fragment feeds 4 WMMAs per K-chunk.
// ---------------------------------------------------------------------------
__global__ __launch_bounds__(256) void qkv_proj(const float* __restrict__ x,
                                                const _Float16* __restrict__ WT,
                                                _Float16* __restrict__ QKV) {
  int wid  = (blockIdx.x * blockDim.x + threadIdx.x) >> 5;  // 0..3071
  int lane = threadIdx.x & 31;
  int r  = lane & 15;
  int hf = lane >> 4;

  int mat   = wid / (MROWS / 16);         // 0..2
  int mtile = wid % (MROWS / 16);         // 0..1023

  const float*    xrow  = x  + (size_t)(mtile * 16 + r) * CDIM;
  const _Float16* wbase = WT + (size_t)mat * HDIM * CDIM;
  const _Float16* w0 = wbase + (size_t)( 0 + r) * CDIM;
  const _Float16* w1 = wbase + (size_t)(16 + r) * CDIM;
  const _Float16* w2 = wbase + (size_t)(32 + r) * CDIM;
  const _Float16* w3 = wbase + (size_t)(48 + r) * CDIM;

  v8f acc0 = {}, acc1 = {}, acc2 = {}, acc3 = {};
  for (int kc = 0; kc < CDIM; kc += 32) {
    // A fragment (16x32 f16): lane row = r; elems j -> K[kc + hf*16 + j]
    // (contiguous order, matching B)
    v16h a;
    const float* pa0 = xrow + kc + hf * 16;
    const float* pa1 = xrow + kc + hf * 16 + 8;
#pragma unroll
    for (int j = 0; j < 8; ++j) {
      a[j]     = (_Float16)pa0[j];
      a[8 + j] = (_Float16)pa1[j];
    }
    // B fragments (32x16 f16): lane col = r; elems j -> K[kc + hf*16 + j]
    int off = kc + hf * 16;
    v16h b0 = *(const v16h*)(w0 + off);
    v16h b1 = *(const v16h*)(w1 + off);
    v16h b2 = *(const v16h*)(w2 + off);
    v16h b3 = *(const v16h*)(w3 + off);
    acc0 = __builtin_amdgcn_wmma_f32_16x16x32_f16(false, a, false, b0, (short)0, acc0, false, false);
    acc1 = __builtin_amdgcn_wmma_f32_16x16x32_f16(false, a, false, b1, (short)0, acc1, false, false);
    acc2 = __builtin_amdgcn_wmma_f32_16x16x32_f16(false, a, false, b2, (short)0, acc2, false, false);
    acc3 = __builtin_amdgcn_wmma_f32_16x16x32_f16(false, a, false, b3, (short)0, acc3, false, false);
  }

  // C fragment: lanes 0-15 -> M=v, lanes 16-31 -> M=v+8; N = r
  _Float16* outp = QKV + (size_t)mat * MROWS * HDIM;
#pragma unroll
  for (int v = 0; v < 8; ++v) {
    size_t row = (size_t)(mtile * 16 + v + 8 * hf) * HDIM;
    outp[row +  0 + r] = (_Float16)acc0[v];
    outp[row + 16 + r] = (_Float16)acc1[v];
    outp[row + 32 + r] = (_Float16)acc2[v];
    outp[row + 48 + r] = (_Float16)acc3[v];
  }
}

// ---------------------------------------------------------------------------
// Kernel 2: flash attention. 4 waves/WG, 64 query rows/WG, 32-key blocks.
// K block staged with GLOBAL_LOAD_ASYNC_TO_LDS_B128 (ASYNCcnt).
// ---------------------------------------------------------------------------
__global__ __launch_bounds__(128) void flash_attn(const _Float16* __restrict__ Q,
                                                  const _Float16* __restrict__ K,
                                                  const _Float16* __restrict__ V,
                                                  float* __restrict__ out) {
  __shared__ _Float16 Ktile[32][64];     // key-major, filled by async DMA
  __shared__ _Float16 VtileT[64][32];    // transposed: [h][key]
  __shared__ float    Stile[4][16][32];
  __shared__ _Float16 Ptile[4][16][32];
  __shared__ float    mArr[4][16], lArr[4][16], cArr[4][16];

  int b     = blockIdx.x >> 6;           // SEQ/64 = 64 tiles per batch
  int qtile = blockIdx.x & 63;
  int tid   = threadIdx.x;
  int wave  = tid >> 5;
  int lane  = tid & 31;
  int r  = lane & 15;
  int hf = lane >> 4;

  const _Float16* Qb = Q + (size_t)b * SEQ * HDIM;
  const _Float16* Kb = K + (size_t)b * SEQ * HDIM;
  const _Float16* Vb = V + (size_t)b * SEQ * HDIM;

  int qbase = qtile * 64;

  // LDS byte offset of Ktile (generic LDS addr low 32 bits == LDS offset)
  uint32_t ktileLds = (uint32_t)(uintptr_t)&Ktile[0][0];

  // Load Q fragments for this wave's 16 rows (resident whole kernel).
  v16h q0, q1;
  {
    const _Float16* qp = Qb + (size_t)(qbase + wave * 16 + r) * HDIM;
#pragma unroll
    for (int j = 0; j < 8; ++j) {
      q0[j]     = qp[hf * 16 + j];
      q0[8 + j] = qp[hf * 16 + 8 + j];
      q1[j]     = qp[32 + hf * 16 + j];
      q1[8 + j] = qp[32 + hf * 16 + 8 + j];
    }
  }

  v8f o0 = {}, o1 = {}, o2 = {}, o3 = {};
  if (lane < 16) {
    mArr[wave][lane] = -__builtin_inff();
    lArr[wave][lane] = 0.0f;
  }

  int waveMaxRow = qbase + wave * 16 + 15;   // last query row owned by wave
  int send = qbase + 64;
  for (int sb = 0; sb < send; sb += 32) {
    __syncthreads();  // previous iteration's LDS reads complete

    // ---- K block: async DMA global -> LDS (no VGPR data path) ----
    {
      // 32 rows x 64 halves = 4 KB = 256 x b128; 2 per thread
      const _Float16* gk = Kb + (size_t)sb * HDIM + tid * 8;
      uint32_t ldsa = ktileLds + (uint32_t)tid * 16u;
      asm volatile("global_load_async_to_lds_b128 %0, %1, off"
                   :: "v"(ldsa), "v"(gk) : "memory");
      asm volatile("global_load_async_to_lds_b128 %0, %1, off"
                   :: "v"(ldsa + 2048u), "v"(gk + 1024) : "memory");
    }
    // ---- V block: load + transpose into LDS ----
    for (int c = tid; c < 256; c += 128) {        // 256 chunks of 8 halves
      int row = c >> 3, col8 = (c & 7) * 8;
      uint4 vd = *(const uint4*)(Vb + (size_t)(sb + row) * HDIM + col8);
      const _Float16* hp = (const _Float16*)&vd;
#pragma unroll
      for (int j = 0; j < 8; ++j) VtileT[col8 + j][row] = hp[j];
    }
    // prefetch next key block into cache
    if (sb + 32 < send) {
      __builtin_prefetch(Kb + (size_t)(sb + 32 + (tid >> 2)) * HDIM, 0, 1);
      __builtin_prefetch(Vb + (size_t)(sb + 32 + (tid >> 2)) * HDIM, 0, 1);
    }
    asm volatile("s_wait_asynccnt 0x0" ::: "memory");
    __syncthreads();

    // Wave-uniform causal skip: block entirely above this wave's rows.
    bool active = (sb <= waveMaxRow);

    if (active) {
      // ---- S = Q * K^T  (16 rows x 32 keys), 4 WMMAs ----
#pragma unroll
      for (int n0 = 0; n0 < 32; n0 += 16) {
        v8f sc = {};
        v16h kb0 = *(const v16h*)&Ktile[n0 + r][hf * 16];        // H chunk 0
        sc = __builtin_amdgcn_wmma_f32_16x16x32_f16(false, q0, false, kb0,
                                                    (short)0, sc, false, false);
        v16h kb1 = *(const v16h*)&Ktile[n0 + r][32 + hf * 16];   // H chunk 1
        sc = __builtin_amdgcn_wmma_f32_16x16x32_f16(false, q1, false, kb1,
                                                    (short)0, sc, false, false);
#pragma unroll
        for (int v = 0; v < 8; ++v)
          Stile[wave][v + 8 * hf][n0 + r] = sc[v];
      }
    }
    __syncthreads();

    if (active && lane < 16) {
      // ---- online softmax: lane < 16 owns query row `lane` of this wave ----
      int   qrow = qbase + wave * 16 + lane;
      float mold = mArr[wave][lane];
      float rmax = -__builtin_inff();
#pragma unroll
      for (int j = 0; j < 32; ++j) {
        float s = Stile[wave][lane][j];
        if (sb + j > qrow) s = -__builtin_inff();   // causal mask
        rmax = fmaxf(rmax, s);
      }
      float mnew = fmaxf(mold, rmax);
      float corr = __expf(mold - mnew);             // 0 on first block
      float sum = 0.0f;
#pragma unroll
      for (int j = 0; j < 32; ++j) {
        float s = Stile[wave][lane][j];
        if (sb + j > qrow) s = -__builtin_inff();
        float p = __expf(s - mnew);
        sum += p;
        Ptile[wave][lane][j] = (_Float16)p;
      }
      mArr[wave][lane] = mnew;
      lArr[wave][lane] = corr * lArr[wave][lane] + sum;
      cArr[wave][lane] = corr;
    }
    __syncthreads();

    if (active) {
      // ---- rescale O and accumulate P * V ----
      float cf[8];
#pragma unroll
      for (int v = 0; v < 8; ++v) cf[v] = cArr[wave][v + 8 * hf];
#pragma unroll
      for (int v = 0; v < 8; ++v) {
        o0[v] *= cf[v]; o1[v] *= cf[v]; o2[v] *= cf[v]; o3[v] *= cf[v];
      }

      // P A-fragment (16x32 f16)
      v16h pa;
#pragma unroll
      for (int j = 0; j < 8; ++j) {
        pa[j]     = Ptile[wave][r][hf * 16 + j];
        pa[8 + j] = Ptile[wave][r][hf * 16 + 8 + j];
      }
      // V B-fragments from transposed tile: col h = n*16+r, elems j -> key hf*16+j
      v16h vb0 = *(const v16h*)&VtileT[ 0 + r][hf * 16];
      v16h vb1 = *(const v16h*)&VtileT[16 + r][hf * 16];
      v16h vb2 = *(const v16h*)&VtileT[32 + r][hf * 16];
      v16h vb3 = *(const v16h*)&VtileT[48 + r][hf * 16];
      o0 = __builtin_amdgcn_wmma_f32_16x16x32_f16(false, pa, false, vb0, (short)0, o0, false, false);
      o1 = __builtin_amdgcn_wmma_f32_16x16x32_f16(false, pa, false, vb1, (short)0, o1, false, false);
      o2 = __builtin_amdgcn_wmma_f32_16x16x32_f16(false, pa, false, vb2, (short)0, o2, false, false);
      o3 = __builtin_amdgcn_wmma_f32_16x16x32_f16(false, pa, false, vb3, (short)0, o3, false, false);
    }
  }

  // ---- final normalize and store fp32 output ----
  float li[8];
#pragma unroll
  for (int v = 0; v < 8; ++v) li[v] = 1.0f / lArr[wave][v + 8 * hf];

  float* ob = out + ((size_t)b * SEQ + qbase + wave * 16) * HDIM;
#pragma unroll
  for (int v = 0; v < 8; ++v) {
    int row = v + 8 * hf;
    ob[row * HDIM +  0 + r] = o0[v] * li[v];
    ob[row * HDIM + 16 + r] = o1[v] * li[v];
    ob[row * HDIM + 32 + r] = o2[v] * li[v];
    ob[row * HDIM + 48 + r] = o3[v] * li[v];
  }
}

// ---------------------------------------------------------------------------
extern "C" void kernel_launch(void* const* d_in, const int* in_sizes, int n_in,
                              void* d_out, int out_size, void* d_ws, size_t ws_size,
                              hipStream_t stream) {
  (void)in_sizes; (void)n_in; (void)out_size; (void)ws_size;
  const float* x  = (const float*)d_in[0];
  const float* Wq = (const float*)d_in[1];
  const float* Wk = (const float*)d_in[2];
  const float* Wv = (const float*)d_in[3];

  char* ws = (char*)d_ws;
  _Float16* WT  = (_Float16*)ws;                                 // 3*64*1024 f16 = 384 KB
  _Float16* QKV = (_Float16*)(ws + 3 * HDIM * CDIM * 2);         // 3*16384*64 f16 = 6 MB
  _Float16* Qf  = QKV;
  _Float16* Kf  = QKV + (size_t)MROWS * HDIM;
  _Float16* Vf  = QKV + (size_t)2 * MROWS * HDIM;

  prep_w<<<(3 * CDIM * HDIM + 255) / 256, 256, 0, stream>>>(Wq, Wk, Wv, WT);

  // 3 matrices * 1024 mtiles = 3072 waves = 384 blocks of 256
  qkv_proj<<<384, 256, 0, stream>>>(x, WT, QKV);

  // B * (SEQ/64) = 256 workgroups, 128 threads (4 waves)
  flash_attn<<<BATCH * (SEQ / 64), 128, 0, stream>>>(Qf, Kf, Vf, (float*)d_out);
}
